// RetouchGenerator_25434796327093
// MI455X (gfx1250) — matrix operand order repaired and checked by reading, not compiled, not run on hardware
//
#include <hip/hip_runtime.h>
#include <hip/hip_bf16.h>

typedef _Float16 v16h __attribute__((ext_vector_type(16)));
typedef _Float16 v8h  __attribute__((ext_vector_type(8)));
typedef float    v8f  __attribute__((ext_vector_type(8)));

union AFrag { v16h v; v8h h[2]; };

#if defined(__gfx1250__) && __has_builtin(__builtin_amdgcn_global_load_async_to_lds_b32) && __has_builtin(__builtin_amdgcn_s_wait_asynccnt)
#define USE_ASYNC_LDS 1
typedef __attribute__((address_space(1))) int* gptr_i32;
typedef __attribute__((address_space(3))) int* lptr_i32;
#else
#define USE_ASYNC_LDS 0
#endif

// ---------------------------------------------------------------------------
// Generic direct 3x3 conv, pad=1, optional stride/relu/bias (small layers)
// ---------------------------------------------------------------------------
__global__ void conv3x3_direct(const float* __restrict__ in,
                               const float* __restrict__ w,
                               const float* __restrict__ bias,
                               float* __restrict__ out,
                               int B, int Cin, int Hin, int Win,
                               int Cout, int Hout, int Wout,
                               int stride, int do_relu, int total) {
  int idx = blockIdx.x * blockDim.x + threadIdx.x;
  if (idx >= total) return;
  int ox = idx % Wout;
  int oy = (idx / Wout) % Hout;
  int oc = (idx / (Wout * Hout)) % Cout;
  int b  = idx / (Wout * Hout * Cout);
  float acc = bias ? bias[oc] : 0.0f;
  int iy0 = oy * stride - 1;
  int ix0 = ox * stride - 1;
  for (int ic = 0; ic < Cin; ++ic) {
    const float* ip = in + ((size_t)(b * Cin + ic) * Hin) * Win;
    const float* wp = w + ((size_t)(oc * Cin + ic)) * 9;
#pragma unroll
    for (int ky = 0; ky < 3; ++ky) {
      int iy = iy0 + ky;
      if (iy < 0 || iy >= Hin) continue;
#pragma unroll
      for (int kx = 0; kx < 3; ++kx) {
        int ix = ix0 + kx;
        if (ix < 0 || ix >= Win) continue;
        acc = fmaf(ip[iy * Win + ix], wp[ky * 3 + kx], acc);
      }
    }
  }
  if (do_relu) acc = fmaxf(acc, 0.0f);
  out[idx] = acc;
}

// ---------------------------------------------------------------------------
// Weight transform for WMMA conv: (oc,ic,3,3) f32 -> [oc][kpos*64+ic] f16
// ---------------------------------------------------------------------------
__global__ void wprep_f16(const float* __restrict__ w, _Float16* __restrict__ wT,
                          int total) {
  int idx = blockIdx.x * blockDim.x + threadIdx.x;
  if (idx >= total) return;
  int r  = idx % 576;       // kpos*64 + ic
  int oc = idx / 576;
  int kpos = r >> 6;
  int ic   = r & 63;
  wT[idx] = (_Float16)w[(size_t)oc * 576 + ic * 9 + kpos];
}

// ---------------------------------------------------------------------------
// 64->64 3x3 conv (stride1 pad1) on 16x16 via WMMA implicit GEMM.
// M=256 spatial, N=64 oc, K=576 ordered (ky*3+kx)*64+ic.
// One block per batch, 8 waves, 64 output tiles of 16x16.
// ---------------------------------------------------------------------------
__global__ void __launch_bounds__(256)
lf_conv_wmma(const float* __restrict__ in,       // (B,64,16,16)
             const _Float16* __restrict__ wT,    // (64,576)
             const float* __restrict__ bias,     // 64 or null
             float* __restrict__ out,            // (B,64,16,16)
             int do_relu) {
  __shared__ alignas(16) _Float16 sIn[18 * 18 * 64];   // HWC f16 with halo
  const int b   = blockIdx.x;
  const int tid = threadIdx.x;

  // Stage padded input, ic-contiguous
  for (int i = tid; i < 18 * 18 * 64; i += 256) {
    int c  = i & 63;
    int sp = i >> 6;
    int y = sp / 18, x = sp % 18;
    float v = 0.0f;
    if (y >= 1 && y <= 16 && x >= 1 && x <= 16)
      v = in[((size_t)(b * 64 + c) << 8) + (y - 1) * 16 + (x - 1)];
    sIn[sp * 64 + c] = (_Float16)v;
  }
  __syncthreads();

  const int wave  = tid >> 5;
  const int lane  = tid & 31;
  const int lhalf = (lane >> 4) & 1;   // 0: lanes 0-15, 1: lanes 16-31
  const int lmod  = lane & 15;

  for (int t = wave; t < 64; t += 8) {
    const int M0 = (t & 15) << 4;
    const int N0 = (t >> 4) << 4;
    const int m  = M0 + lmod;
    const int oy = m >> 4, ox = m & 15;
    const int n  = N0 + lmod;

    v8f acc = {};
    for (int k0 = 0; k0 < 576; k0 += 32) {
      AFrag a;
#pragma unroll
      for (int s = 0; s < 2; ++s) {
        int kb   = k0 + lhalf * 8 + s * 16;
        int kpos = kb >> 6;
        int ic0  = kb & 63;
        int ky = kpos / 3, kx = kpos - ky * 3;
        int addr = ((oy + ky) * 18 + (ox + kx)) * 64 + ic0;
        a.h[s] = *reinterpret_cast<const v8h*>(&sIn[addr]);
      }
      AFrag bf;
      const _Float16* wp = wT + (size_t)n * 576 + k0 + lhalf * 8;
      bf.h[0] = *reinterpret_cast<const v8h*>(wp);
      bf.h[1] = *reinterpret_cast<const v8h*>(wp + 16);

      acc = __builtin_amdgcn_wmma_f32_16x16x32_f16(
          false, a.v, false, bf.v, (short)0, acc, false, false);
    }

    const float bv = bias ? bias[n] : 0.0f;
#pragma unroll
    for (int v = 0; v < 8; ++v) {
      int row = M0 + v + lhalf * 8;           // D layout: lanes>=16 hold M=v+8
      float r = acc[v] + bv;
      if (do_relu) r = fmaxf(r, 0.0f);
      out[((size_t)(b * 64 + n) << 8) + row] = r;
    }
  }
}

// ---------------------------------------------------------------------------
// FC: out[b][n] = relu?(sum_k in[b][k]*w[n][k] + bias[n])
// ---------------------------------------------------------------------------
__global__ void fc_layer(const float* __restrict__ in, const float* __restrict__ w,
                         const float* __restrict__ bias, float* __restrict__ out,
                         int B, int K, int N, int do_relu) {
  int idx = blockIdx.x * blockDim.x + threadIdx.x;
  if (idx >= B * N) return;
  int n = idx % N, b = idx / N;
  const float* ip = in + (size_t)b * K;
  const float* wp = w + (size_t)n * K;
  float acc = bias[n];
  for (int k = 0; k < K; ++k) acc = fmaf(ip[k], wp[k], acc);
  if (do_relu) acc = fmaxf(acc, 0.0f);
  out[idx] = acc;
}

// ---------------------------------------------------------------------------
// fusion = relu(gf[b,ic] + lf[b,ic,m]); grid = 1x1 conv (96x64) + bias
// ---------------------------------------------------------------------------
__global__ void fusion_pred(const float* __restrict__ lf, const float* __restrict__ gf,
                            const float* __restrict__ pw, const float* __restrict__ pb,
                            float* __restrict__ grid, int total) {
  int idx = blockIdx.x * blockDim.x + threadIdx.x;
  if (idx >= total) return;
  int m  = idx & 255;
  int oc = (idx >> 8) % 96;
  int b  = idx / (96 * 256);
  const float* lfp = lf + ((size_t)b * 64 << 8);
  const float* gfp = gf + (size_t)b * 64;
  const float* wp  = pw + (size_t)oc * 64;
  float acc = pb[oc];
  for (int ic = 0; ic < 64; ++ic) {
    float f = fmaxf(gfp[ic] + lfp[(ic << 8) + m], 0.0f);
    acc = fmaf(f, wp[ic], acc);
  }
  grid[idx] = acc;
}

// ---------------------------------------------------------------------------
// Guide + trilinear slice + per-pixel affine. One block = 32x32 pixel tile
// (shares the 4 spatial grid corners). Grid corners staged into LDS via
// async-to-LDS (ASYNCcnt) when available. Each thread owns 4 consecutive
// pixels of one row -> all high-res I/O is b128.
// high: (2,3,1024,1024), grid: (2,96,256) [c*8+z major], out: (2,3,1024,1024)
// ---------------------------------------------------------------------------
__global__ void __launch_bounds__(256)
guide_slice(const float* __restrict__ high, const float* __restrict__ grid,
            const float* __restrict__ pw_mat, const float* __restrict__ pw_bias,
            const float* __restrict__ pw_bias_tag,
            const float* __restrict__ rho_a, const float* __restrict__ rho_t,
            float* __restrict__ out) {
  __shared__ alignas(16) float sg[4 * 8 * 12];   // [corner][z][ch]
  __shared__ float s_rt[48], s_ra[48];
  const int tid = threadIdx.x;
  const int blk = blockIdx.x;
  const int b   = blk >> 10;          // 1024 tiles per image
  const int tYX = blk & 1023;
  const int tY = tYX >> 5, tX = tYX & 31;
  const int x0 = tX << 5, y0 = tY << 5;

  // Tile-uniform spatial corners (constant over a 32px half-cell)
  const float fxf = fmaxf(floorf((x0 + 0.5f) * 0.015625f - 0.5f), 0.0f);
  const float fyf = fmaxf(floorf((y0 + 0.5f) * 0.015625f - 0.5f), 0.0f);
  const int fx = (int)fxf, fy = (int)fyf;
  const int cx = min(fx + 1, 15), cy = min(fy + 1, 15);

  if (tid < 48) { s_rt[tid] = rho_t[tid]; s_ra[tid] = rho_a[tid]; }

  // Stage grid corners: sg[corner*96 + z*12 + c]
  {
    int i = tid;
    if (i < 384) {
      int corner = i / 96;
      int rem = i - corner * 96;
      int z = rem / 12, c = rem - z * 12;
      int yy = (corner & 2) ? cy : fy;
      int xx = (corner & 1) ? cx : fx;
      size_t gidx = (size_t)b * 24576 + (size_t)(c * 8 + z) * 256 + yy * 16 + xx;
#if USE_ASYNC_LDS
      __builtin_amdgcn_global_load_async_to_lds_b32(
          (gptr_i32)(grid + gidx), (lptr_i32)&sg[i], 0, 0);
#else
      sg[i] = grid[gidx];
#endif
    }
#if USE_ASYNC_LDS
    __builtin_amdgcn_s_wait_asynccnt(0);
#endif
  }
  __syncthreads();

  float pm[9], bt[3];
#pragma unroll
  for (int i = 0; i < 9; ++i) pm[i] = pw_mat[i];
#pragma unroll
  for (int i = 0; i < 3; ++i) bt[i] = pw_bias_tag[i];
  const float pwb = pw_bias[0];

  const size_t imgBase = (size_t)b * 3u * 1048576u;

  // Each thread: 4 consecutive pixels in one row (b128 loads/stores)
  const int py  = tid >> 3;
  const int px4 = (tid & 7) << 2;
  const int x = x0 + px4;
  const int y = y0 + py;
  const size_t base = imgBase + (size_t)y * 1024 + x;

  const float4 r4 = *reinterpret_cast<const float4*>(&high[base]);
  const float4 g4 = *reinterpret_cast<const float4*>(&high[base + 1048576]);
  const float4 b4 = *reinterpret_cast<const float4*>(&high[base + 2097152]);
  __builtin_prefetch(&high[base + 2097152 + 4096], 0, 1);

  const float rr[4] = {r4.x, r4.y, r4.z, r4.w};
  const float gg[4] = {g4.x, g4.y, g4.z, g4.w};
  const float bb[4] = {b4.x, b4.y, b4.z, b4.w};

  const float wy = (y + 0.5f) * 0.015625f - 0.5f - fyf;
  float4 o0, o1, o2;
  float* o0p = &o0.x; float* o1p = &o1.x; float* o2p = &o2.x;

#pragma unroll
  for (int j = 0; j < 4; ++j) {
    const float r = rr[j], g = gg[j], bl = bb[j];

    // learned pointwise color curve -> guide
    float gv0 = fmaf(pm[0], r, fmaf(pm[1], g, fmaf(pm[2], bl, bt[0])));
    float gv1 = fmaf(pm[3], r, fmaf(pm[4], g, fmaf(pm[5], bl, bt[1])));
    float gv2 = fmaf(pm[6], r, fmaf(pm[7], g, fmaf(pm[8], bl, bt[2])));
    float guide = pwb;
#pragma unroll
    for (int k = 0; k < 16; ++k) {
      float t0 = fmaxf(gv0 - s_rt[k * 3 + 0], 0.0f);
      float t1 = fmaxf(gv1 - s_rt[k * 3 + 1], 0.0f);
      float t2 = fmaxf(gv2 - s_rt[k * 3 + 2], 0.0f);
      guide = fmaf(t0, s_ra[k * 3 + 0],
              fmaf(t1, s_ra[k * 3 + 1],
              fmaf(t2, s_ra[k * 3 + 2], guide)));
    }

    // z coordinate
    float gz  = fminf(fmaxf(guide, 0.0f), 1.0f) * 8.0f;
    float fzf = fmaxf(floorf(gz - 0.5f), 0.0f);
    float wz  = fabsf(gz - 0.5f - fzf);
    int fz = (int)fzf;
    int cz = min(fz + 1, 7);

    float wx = (x + j + 0.5f) * 0.015625f - 0.5f - fxf;
    float cw[4];
    cw[0] = (1.0f - wx) * (1.0f - wy);
    cw[1] = wx * (1.0f - wy);
    cw[2] = (1.0f - wx) * wy;
    cw[3] = wx * wy;

    float a[12];
#pragma unroll
    for (int c = 0; c < 12; ++c) a[c] = 0.0f;
#pragma unroll
    for (int corner = 0; corner < 4; ++corner) {
      const float4* pf = reinterpret_cast<const float4*>(&sg[corner * 96 + fz * 12]);
      const float4* pc = reinterpret_cast<const float4*>(&sg[corner * 96 + cz * 12]);
      float wA = cw[corner] * (1.0f - wz);
      float wB = cw[corner] * wz;
#pragma unroll
      for (int q = 0; q < 3; ++q) {
        float4 vf = pf[q];
        float4 vc = pc[q];
        a[q * 4 + 0] = fmaf(wA, vf.x, fmaf(wB, vc.x, a[q * 4 + 0]));
        a[q * 4 + 1] = fmaf(wA, vf.y, fmaf(wB, vc.y, a[q * 4 + 1]));
        a[q * 4 + 2] = fmaf(wA, vf.z, fmaf(wB, vc.z, a[q * 4 + 2]));
        a[q * 4 + 3] = fmaf(wA, vf.w, fmaf(wB, vc.w, a[q * 4 + 3]));
      }
    }

    o0p[j] = fmaf(a[0], r, fmaf(a[1],  g, fmaf(a[2],  bl, a[3])));
    o1p[j] = fmaf(a[4], r, fmaf(a[5],  g, fmaf(a[6],  bl, a[7])));
    o2p[j] = fmaf(a[8], r, fmaf(a[9],  g, fmaf(a[10], bl, a[11])));
  }

  *reinterpret_cast<float4*>(&out[base])           = o0;
  *reinterpret_cast<float4*>(&out[base + 1048576]) = o1;
  *reinterpret_cast<float4*>(&out[base + 2097152]) = o2;
}

// ---------------------------------------------------------------------------
// Host launcher
// ---------------------------------------------------------------------------
static inline size_t alignup256(size_t x) { return (x + 255) & ~(size_t)255; }

extern "C" void kernel_launch(void* const* d_in, const int* in_sizes, int n_in,
                              void* d_out, int out_size, void* d_ws, size_t ws_size,
                              hipStream_t stream) {
  (void)in_sizes; (void)n_in; (void)out_size; (void)ws_size;
  const float* high   = (const float*)d_in[0];
  const float* low    = (const float*)d_in[1];
  const float* ll1w = (const float*)d_in[2];  const float* ll1b = (const float*)d_in[3];
  const float* ll2w = (const float*)d_in[4];  const float* ll2b = (const float*)d_in[5];
  const float* ll3w = (const float*)d_in[6];  const float* ll3b = (const float*)d_in[7];
  const float* ll4w = (const float*)d_in[8];  const float* ll4b = (const float*)d_in[9];
  const float* lf1w = (const float*)d_in[10]; const float* lf1b = (const float*)d_in[11];
  const float* lf2w = (const float*)d_in[12];
  const float* gf1w = (const float*)d_in[13]; const float* gf1b = (const float*)d_in[14];
  const float* gf2w = (const float*)d_in[15]; const float* gf2b = (const float*)d_in[16];
  const float* fc1w = (const float*)d_in[17]; const float* fc1b = (const float*)d_in[18];
  const float* fc2w = (const float*)d_in[19]; const float* fc2b = (const float*)d_in[20];
  const float* fc3w = (const float*)d_in[21]; const float* fc3b = (const float*)d_in[22];
  const float* predw = (const float*)d_in[23]; const float* predb = (const float*)d_in[24];
  const float* pw_mat = (const float*)d_in[25];
  const float* pw_bias = (const float*)d_in[26];
  const float* pw_bias_tag = (const float*)d_in[27];
  const float* rho_a = (const float*)d_in[28];
  const float* rho_t = (const float*)d_in[29];

  char* ws = (char*)d_ws;
  size_t off = 0;
  auto alloc = [&](size_t bytes) { size_t o = off; off = alignup256(off + bytes); return o; };
  float* x1   = (float*)(ws + alloc(2 * 8  * 128 * 128 * 4));
  float* x2   = (float*)(ws + alloc(2 * 16 * 64 * 64 * 4));
  float* x3   = (float*)(ws + alloc(2 * 32 * 32 * 32 * 4));
  float* x4   = (float*)(ws + alloc(2 * 64 * 16 * 16 * 4));
  float* lf1o = (float*)(ws + alloc(2 * 64 * 256 * 4));
  float* lf2o = (float*)(ws + alloc(2 * 64 * 256 * 4));
  float* gf1o = (float*)(ws + alloc(2 * 64 * 64 * 4));
  float* gf2o = (float*)(ws + alloc(2 * 64 * 16 * 4));
  float* fc1o = (float*)(ws + alloc(2 * 256 * 4));
  float* fc2o = (float*)(ws + alloc(2 * 128 * 4));
  float* fc3o = (float*)(ws + alloc(2 * 64 * 4));
  float* gridb = (float*)(ws + alloc(2 * 96 * 256 * 4));
  _Float16* wt1 = (_Float16*)(ws + alloc(64 * 576 * 2));
  _Float16* wt2 = (_Float16*)(ws + alloc(64 * 576 * 2));

  auto nb = [](int n) { return (n + 255) / 256; };

  // WMMA weight transforms
  wprep_f16<<<nb(36864), 256, 0, stream>>>(lf1w, wt1, 36864);
  wprep_f16<<<nb(36864), 256, 0, stream>>>(lf2w, wt2, 36864);

  // Low-level feature ladder (stride-2 3x3 convs)
  conv3x3_direct<<<nb(262144), 256, 0, stream>>>(low, ll1w, ll1b, x1, 2, 3, 256, 256, 8, 128, 128, 2, 1, 262144);
  conv3x3_direct<<<nb(131072), 256, 0, stream>>>(x1, ll2w, ll2b, x2, 2, 8, 128, 128, 16, 64, 64, 2, 1, 131072);
  conv3x3_direct<<<nb(65536),  256, 0, stream>>>(x2, ll3w, ll3b, x3, 2, 16, 64, 64, 32, 32, 32, 2, 1, 65536);
  conv3x3_direct<<<nb(32768),  256, 0, stream>>>(x3, ll4w, ll4b, x4, 2, 32, 32, 32, 64, 16, 16, 2, 1, 32768);

  // Local features via WMMA implicit GEMM
  lf_conv_wmma<<<2, 256, 0, stream>>>(x4, wt1, lf1b, lf1o, 1);
  lf_conv_wmma<<<2, 256, 0, stream>>>(lf1o, wt2, nullptr, lf2o, 0);

  // Global feature path
  conv3x3_direct<<<nb(8192), 256, 0, stream>>>(x4, gf1w, gf1b, gf1o, 2, 64, 16, 16, 64, 8, 8, 2, 1, 8192);
  conv3x3_direct<<<nb(2048), 256, 0, stream>>>(gf1o, gf2w, gf2b, gf2o, 2, 64, 8, 8, 64, 4, 4, 2, 1, 2048);
  fc_layer<<<nb(512), 256, 0, stream>>>(gf2o, fc1w, fc1b, fc1o, 2, 1024, 256, 1);
  fc_layer<<<nb(256), 256, 0, stream>>>(fc1o, fc2w, fc2b, fc2o, 2, 256, 128, 1);
  fc_layer<<<nb(128), 256, 0, stream>>>(fc2o, fc3w, fc3b, fc3o, 2, 128, 64, 0);

  // Fusion + 1x1 pred conv -> bilateral grid (2,96,256)
  fusion_pred<<<nb(49152), 256, 0, stream>>>(lf2o, fc3o, predw, predb, gridb, 49152);

  // Guide + trilinear slice + per-pixel affine (memory-bound main kernel)
  guide_slice<<<2048, 256, 0, stream>>>(high, gridb, pw_mat, pw_bias, pw_bias_tag,
                                        rho_a, rho_t, (float*)d_out);
}
